// _CNNSelfAttention_12000138625643
// MI455X (gfx1250) — matrix operand order repaired and checked by reading, not compiled, not run on hardware
//
#include <hip/hip_runtime.h>

#define BATCH 4
#define CDIM  256
#define NDIM  4096   // 64*64 spatial positions
#define BC    32     // key/value block for flash attention

typedef __bf16 bf16_t;
typedef __attribute__((ext_vector_type(16))) __bf16 v16bf;
typedef __attribute__((ext_vector_type(8)))  float  v8f;

union FragU {
  v16bf v;
  uint4 q[2];
};

__device__ __forceinline__ v8f wmma_bf16(v16bf a, v16bf b, v8f c) {
  // D(16x16 f32) = A(16x32 bf16) * B(32x16 bf16) + C
  return __builtin_amdgcn_wmma_f32_16x16x32_bf16(
      /*neg_a=*/false, a, /*neg_b=*/false, b,
      /*c_mod=*/(short)0, c, /*reuse_a=*/false, /*reuse_b=*/false);
}

// ---------------------------------------------------------------------------
// Tensor Data Mover staging (gfx1250): one 2D-tile DMA per K/V tile, LDS
// row padding applied by the TDM itself. Gated on builtin availability.
// ---------------------------------------------------------------------------
#if defined(__gfx1250__) && __has_builtin(__builtin_amdgcn_tensor_load_to_lds)
#define USE_TDM 1
#else
#define USE_TDM 0
#endif

#if USE_TDM
#if __has_include(<hip/amd_detail/amd_gfx1250_TDM.h>)
#define TDM_SIX 1    // therock-10.0 headers -> 6-arg builtin
#else
#define TDM_SIX 0    // ROCm 7.2 -> 5-arg builtin
#endif

typedef unsigned int u32x4 __attribute__((ext_vector_type(4)));
typedef int          i32x4 __attribute__((ext_vector_type(4)));
typedef int          i32x8 __attribute__((ext_vector_type(8)));

typedef __attribute__((address_space(3))) const void lds_cvoid_t;

__device__ __forceinline__ unsigned lds_off32(const void* p) {
  return (unsigned)(unsigned long long)(lds_cvoid_t*)p;   // addrspacecast -> LDS byte offset
}

// 2D tile load: tile_d1 rows of tile_d0 bf16 elements, row stride
// stride0_elems, with LDS padding of (pad_amount_code+1) DWORDs every
// 2^(pad_interval_code+1) DWORDs.
__device__ __forceinline__ void tdm_load_2d(
    unsigned lds_addr, const void* gptr,
    unsigned tensor_d0, unsigned tensor_d1,
    unsigned tile_d0, unsigned tile_d1,
    unsigned long long stride0_elems,
    unsigned pad_interval_code, unsigned pad_amount_code)
{
  const unsigned long long ga = (unsigned long long)gptr;
  u32x4 g0;
  g0[0] = 1u;                                              // count=1 (valid user D#)
  g0[1] = lds_addr;                                        // [63:32] lds_addr
  g0[2] = (unsigned)(ga & 0xffffffffull);                  // [95:64] global_addr lo
  g0[3] = (unsigned)((ga >> 32) & 0x01ffffffull)           // [120:96] global_addr hi
        | (2u << 30);                                      // [127:126] type=2 (image)

  unsigned w[8] = {0u, 0u, 0u, 0u, 0u, 0u, 0u, 0u};
  w[0] = (1u << 16)                                        // data_size = 2 bytes
       | (1u << 20)                                        // pad_enable
       | (pad_interval_code << 22)
       | (pad_amount_code << 25);
  w[1] = (tensor_d0 & 0xffffu) << 16;                      // tensor_dim0[15:0]  @ [63:48]
  w[2] = ((tensor_d0 >> 16) & 0xffffu)                     // tensor_dim0[31:16] @ [79:64]
       | ((tensor_d1 & 0xffffu) << 16);                    // tensor_dim1[15:0]  @ [95:80]
  w[3] = ((tensor_d1 >> 16) & 0xffffu)                     // tensor_dim1[31:16] @ [111:96]
       | ((tile_d0 & 0xffffu) << 16);                      // tile_dim0          @ [127:112]
  w[4] = (tile_d1 & 0xffffu);                              // tile_dim1          @ [143:128]
  w[5] = (unsigned)(stride0_elems & 0xffffffffull);        // dim0_stride lo     @ [191:160]
  w[6] = (unsigned)((stride0_elems >> 32) & 0xffffull);    // dim0_stride hi     @ [207:192]

  i32x8 g1;
  #pragma unroll
  for (int i = 0; i < 8; ++i) g1[i] = (int)w[i];
  i32x4 z4 = {0, 0, 0, 0};
#if TDM_SIX
  i32x8 z8 = {0, 0, 0, 0, 0, 0, 0, 0};
  __builtin_amdgcn_tensor_load_to_lds(g0, g1, z4, z4, z8, 0);
#else
  __builtin_amdgcn_tensor_load_to_lds(g0, g1, z4, z4, 0);
#endif
}
#endif  // USE_TDM

// ---------------------------------------------------------------------------
// Kernel 1: 1x1-conv QKV projection via WMMA (two interleaved acc chains).
// ---------------------------------------------------------------------------
template <bool VLAYOUT>
__device__ __forceinline__ void project_one(
    const v16bf (&af)[8], const float* __restrict__ W,
    const float* __restrict__ bp, bf16_t* __restrict__ dst,
    int b, int n0, int lh, int l16)
{
  #pragma unroll 1
  for (int ot = 0; ot < 16; ot += 2) {
    const int oc0 = ot * 16 + l16;
    const int oc1 = oc0 + 16;
    const float b0 = bp[oc0];
    const float b1 = bp[oc1];
    v8f a0, a1;
    #pragma unroll
    for (int r = 0; r < 8; ++r) { a0[r] = b0; a1[r] = b1; }
    #pragma unroll
    for (int ks = 0; ks < 8; ++ks) {
      // B layout: lane -> column oc; element e -> K = 16*lh + e (contiguous row)
      const float* w0 = W + (size_t)oc0 * CDIM + ks * 32 + lh * 16;
      const float* w1 = W + (size_t)oc1 * CDIM + ks * 32 + lh * 16;
      v16bf wb0, wb1;
      #pragma unroll
      for (int e = 0; e < 16; ++e) { wb0[e] = (bf16_t)w0[e]; wb1[e] = (bf16_t)w1[e]; }
      a0 = wmma_bf16(af[ks], wb0, a0);     // two independent chains ->
      a1 = wmma_bf16(af[ks], wb1, a1);     // no back-to-back RAW hazard
    }
    #pragma unroll
    for (int r = 0; r < 8; ++r) {
      const int n = n0 + r + lh * 8;       // D: row = r + 8*lh, col = l16
      const bf16_t h0 = (bf16_t)a0[r];
      const bf16_t h1 = (bf16_t)a1[r];
      if (!VLAYOUT) {
        dst[((size_t)b * NDIM + n) * CDIM + oc0] = h0;
        dst[((size_t)b * NDIM + n) * CDIM + oc1] = h1;
      } else {
        dst[((size_t)b * CDIM + oc0) * NDIM + n] = h0;
        dst[((size_t)b * CDIM + oc1) * NDIM + n] = h1;
      }
    }
  }
}

__global__ __launch_bounds__(256) void qkv_project(
    const float* __restrict__ x,
    const float* __restrict__ Wq, const float* __restrict__ Wk,
    const float* __restrict__ Wv,
    const float* __restrict__ bq, const float* __restrict__ bk,
    const float* __restrict__ bv,
    bf16_t* __restrict__ Qt, bf16_t* __restrict__ Kt, bf16_t* __restrict__ Vc)
{
  const int lane = threadIdx.x & 31;
  const int wave = threadIdx.x >> 5;
  const int lh   = lane >> 4;
  const int l16  = lane & 15;

  const int bpB = NDIM / 128;
  const int b   = blockIdx.x / bpB;
  const int n0  = (blockIdx.x % bpB) * 128 + wave * 16;

  const float* xb  = x + (size_t)b * CDIM * NDIM;
  const int   arow = n0 + l16;

  // A fragments: X^T rows n0..n0+15; A layout K = e + 16*lh.
  v16bf af[8];
  #pragma unroll
  for (int ks = 0; ks < 8; ++ks) {
    #pragma unroll
    for (int e = 0; e < 16; ++e) {
      const int c = ks * 32 + e + (lh << 4);
      af[ks][e] = (bf16_t)xb[(size_t)c * NDIM + arow];
    }
  }

  project_one<false>(af, Wq, bq, Qt, b, n0, lh, l16);
  project_one<false>(af, Wk, bk, Kt, b, n0, lh, l16);
  project_one<true >(af, Wv, bv, Vc, b, n0, lh, l16);
}

// ---------------------------------------------------------------------------
// Kernel 2: flash attention. 128 threads = 4 wave32s; each wave owns 16
// query rows. K/V tiles staged to LDS by the Tensor Data Mover when present.
// ---------------------------------------------------------------------------
__global__ __launch_bounds__(128) void flash_attn(
    const bf16_t* __restrict__ Qt, const bf16_t* __restrict__ Kt,
    const bf16_t* __restrict__ Vc, float* __restrict__ out)
{
  __shared__ bf16_t sK[BC][CDIM + 16];    // keys,   [m][c], row = 544 B
  __shared__ bf16_t sV[CDIM][BC + 8];     // values, [c][m], row =  80 B
  __shared__ bf16_t sP[4][16][BC + 8];    // per-wave P tile

  const int tid  = threadIdx.x;
  const int lane = tid & 31;
  const int wave = tid >> 5;
  const int lh   = lane >> 4;
  const int l16  = lane & 15;

  const int bpB = NDIM / 64;
  const int b   = blockIdx.x / bpB;
  const int n0  = (blockIdx.x % bpB) * 64 + wave * 16;

  const bf16_t* Kb = Kt + (size_t)b * NDIM * CDIM;
  const bf16_t* Vb = Vc + (size_t)b * CDIM * NDIM;

  // Q fragments resident in registers for the whole pass.
  v16bf qf[8];
  {
    const bf16_t* qrow = Qt + ((size_t)b * NDIM + n0 + l16) * CDIM;
    #pragma unroll
    for (int ks = 0; ks < 8; ++ks) {
      FragU f;
      f.q[0] = *(const uint4*)(qrow + ks * 32 + lh * 16);
      f.q[1] = *(const uint4*)(qrow + ks * 32 + lh * 16 + 8);
      qf[ks] = f.v;
    }
  }

  v8f oacc[16];
  #pragma unroll
  for (int t = 0; t < 16; ++t) {
    #pragma unroll
    for (int r = 0; r < 8; ++r) oacc[t][r] = 0.f;
  }
  float rmax[8], rsum[8];
  #pragma unroll
  for (int r = 0; r < 8; ++r) { rmax[r] = -1e30f; rsum[r] = 0.f; }

  for (int m0 = 0; m0 < NDIM; m0 += BC) {
    __syncthreads();   // previous iteration's LDS reads complete
#if USE_TDM
    if (wave == 0) {
      // K tile: BC rows x CDIM bf16; pad 8 DWORDs every 128 DWORDs -> 272-half rows
      tdm_load_2d(lds_off32(&sK[0][0]), Kb + (size_t)m0 * CDIM,
                  CDIM, NDIM, CDIM, BC, CDIM, 6u, 7u);
      // V tile: CDIM rows x BC bf16; pad 4 DWORDs every 16 DWORDs -> 40-half rows
      tdm_load_2d(lds_off32(&sV[0][0]), Vb + m0,
                  NDIM, CDIM, BC, CDIM, NDIM, 3u, 3u);
      __builtin_amdgcn_s_wait_tensorcnt(0);
    }
#else
    for (int i = tid; i < BC * (CDIM / 8); i += 128) {
      const int r = i >> 5, cc = i & 31;
      *(uint4*)&sK[r][cc * 8] =
          *(const uint4*)(Kb + ((size_t)(m0 + r)) * CDIM + cc * 8);
    }
    for (int i = tid; i < CDIM * (BC / 8); i += 128) {
      const int c = i >> 2, cc = i & 3;
      *(uint4*)&sV[c][cc * 8] =
          *(const uint4*)(Vb + (size_t)c * NDIM + m0 + cc * 8);
    }
    if (m0 + BC < NDIM) {
      __builtin_prefetch(Kb + ((size_t)(m0 + BC)) * CDIM + (size_t)tid * 64, 0, 1);
      __builtin_prefetch(Vb + (size_t)(tid * 2) * NDIM + m0 + BC, 0, 1);
    }
#endif
    __syncthreads();

    // S = Q(16x256) * K-block(256xBC); interleave the two sub-tile chains.
    v8f st0, st1;
    #pragma unroll
    for (int r = 0; r < 8; ++r) { st0[r] = 0.f; st1[r] = 0.f; }
    #pragma unroll
    for (int ks = 0; ks < 8; ++ks) {
      FragU k0, k1;
      k0.q[0] = *(const uint4*)&sK[l16][ks * 32 + lh * 16];
      k0.q[1] = *(const uint4*)&sK[l16][ks * 32 + lh * 16 + 8];
      k1.q[0] = *(const uint4*)&sK[16 + l16][ks * 32 + lh * 16];
      k1.q[1] = *(const uint4*)&sK[16 + l16][ks * 32 + lh * 16 + 8];
      st0 = wmma_bf16(qf[ks], k0.v, st0);
      st1 = wmma_bf16(qf[ks], k1.v, st1);
    }

    // Online softmax; row stats per-lane (row = r + 8*lh).
    float tm[8];
    #pragma unroll
    for (int r = 0; r < 8; ++r) tm[r] = fmaxf(st0[r], st1[r]);
    #pragma unroll
    for (int off = 8; off >= 1; off >>= 1) {
      #pragma unroll
      for (int r = 0; r < 8; ++r)
        tm[r] = fmaxf(tm[r], __shfl_xor(tm[r], off, 32));
    }

    float alpha[8], ps[8];
    #pragma unroll
    for (int r = 0; r < 8; ++r) {
      const float nm = fmaxf(rmax[r], tm[r]);
      alpha[r] = __expf(rmax[r] - nm);
      rmax[r]  = nm;
    }
    #pragma unroll
    for (int r = 0; r < 8; ++r) {
      const float p0 = __expf(st0[r] - rmax[r]);
      const float p1 = __expf(st1[r] - rmax[r]);
      ps[r] = p0 + p1;
      sP[wave][r + lh * 8][l16]      = (bf16_t)p0;
      sP[wave][r + lh * 8][16 + l16] = (bf16_t)p1;
    }
    #pragma unroll
    for (int off = 8; off >= 1; off >>= 1) {
      #pragma unroll
      for (int r = 0; r < 8; ++r) ps[r] += __shfl_xor(ps[r], off, 32);
    }
    #pragma unroll
    for (int r = 0; r < 8; ++r) rsum[r] = rsum[r] * alpha[r] + ps[r];

    #pragma unroll
    for (int t = 0; t < 16; ++t) {
      #pragma unroll
      for (int r = 0; r < 8; ++r) oacc[t][r] *= alpha[r];
    }

    __syncthreads();   // cross-lane sP visibility before A-fragment reads

    // O += P(16xBC) * V-block(BCx16): 16 independent accumulator chains.
    FragU pa;          // A layout: row = l16, K = (e&7) + 16*(e>>3) + 8*lh
    pa.q[0] = *(const uint4*)&sP[wave][l16][lh * 8];
    pa.q[1] = *(const uint4*)&sP[wave][l16][lh * 8 + 16];
    #pragma unroll
    for (int ct = 0; ct < 16; ++ct) {
      const int cr = ct * 16 + l16;
      FragU vbf;
      vbf.q[0] = *(const uint4*)&sV[cr][lh * 8];
      vbf.q[1] = *(const uint4*)&sV[cr][lh * 8 + 16];
      oacc[ct] = wmma_bf16(pa.v, vbf.v, oacc[ct]);
    }
  }

  // Epilogue: O / rowsum -> out[b][c][n] (f32)
  float inv[8];
  #pragma unroll
  for (int r = 0; r < 8; ++r) inv[r] = 1.0f / rsum[r];
  float* ob = out + (size_t)b * CDIM * NDIM;
  #pragma unroll 1
  for (int ct = 0; ct < 16; ++ct) {
    const int c = ct * 16 + l16;
    #pragma unroll
    for (int r = 0; r < 8; ++r) {
      const int n = n0 + r + lh * 8;
      ob[(size_t)c * NDIM + n] = oacc[ct][r] * inv[r];
    }
  }
}

// ---------------------------------------------------------------------------
extern "C" void kernel_launch(void* const* d_in, const int* in_sizes, int n_in,
                              void* d_out, int out_size, void* d_ws, size_t ws_size,
                              hipStream_t stream) {
  (void)in_sizes; (void)n_in; (void)out_size; (void)ws_size;
  const float* x  = (const float*)d_in[0];
  const float* Wq = (const float*)d_in[1];
  const float* Wk = (const float*)d_in[2];
  const float* Wv = (const float*)d_in[3];
  const float* bq = (const float*)d_in[4];
  const float* bk = (const float*)d_in[5];
  const float* bv = (const float*)d_in[6];
  float* out = (float*)d_out;

  bf16_t* Qt = (bf16_t*)d_ws;                       // [B][N][C] bf16
  bf16_t* Kt = Qt + (size_t)BATCH * NDIM * CDIM;    // [B][N][C] bf16
  bf16_t* Vc = Kt + (size_t)BATCH * NDIM * CDIM;    // [B][C][N] bf16

  qkv_project<<<dim3(BATCH * (NDIM / 128)), dim3(256), 0, stream>>>(
      x, Wq, Wk, Wv, bq, bk, bv, Qt, Kt, Vc);
  flash_attn<<<dim3(BATCH * (NDIM / 64)), dim3(128), 0, stream>>>(
      Qt, Kt, Vc, out);
}
